// EdgeFeatureGCN_11940009083187
// MI455X (gfx1250) — matrix operand order repaired and checked by reading, not compiled, not run on hardware
//
#include <hip/hip_runtime.h>

// ---------------------------------------------------------------------------
// EdgeFeatureGCN on MI455X (gfx1250): bf16 WMMA (f32 accumulate) for all
// K>=32 GEMMs; fully fused edge head (1 block = 16 edges, all intermediates
// in LDS); f32 atomic scatter for the GCN aggregation.
// ---------------------------------------------------------------------------

#define NN 50000      // nodes
#define NE 800000     // edges
#define HD 128        // hidden dim

typedef __bf16 v16bf __attribute__((ext_vector_type(16)));
typedef float  v8f   __attribute__((ext_vector_type(8)));

union FragBF { unsigned int u[8]; v16bf v; };

__device__ __forceinline__ unsigned bf16u(float x) {
    unsigned u = __float_as_uint(x);
    return (u + 0x7FFFu + ((u >> 16) & 1u)) >> 16;   // RNE f32->bf16
}
__device__ __forceinline__ float bf16f(unsigned short h) {
    return __uint_as_float(((unsigned)h) << 16);
}
__device__ __forceinline__ unsigned pack2(float lo, float hi) {
    return bf16u(lo) | (bf16u(hi) << 16);
}

// A-fragment (16x32 bf16 tile, row-major [16][ldk] in LDS).
// ISA 7.12.2: lanes0-15 -> M=lane, K=0..7 (v0..3), 16..23 (v4..7);
//             lanes16-31 -> +8 on K.
__device__ __forceinline__ v16bf afrag_lds(const unsigned short* sA, int ldk,
                                           int kbase, int lane) {
    FragBF f;
    int m  = lane & 15;
    int kh = ((lane >> 4) & 1) << 3;
    const unsigned* p = (const unsigned*)(sA + m * ldk + kbase + kh);
#pragma unroll
    for (int j = 0; j < 8; ++j) {
        int k = 2 * (j & 3) + ((j >> 2) << 4);
        f.u[j] = p[k >> 1];
    }
    return f.v;
}

// B-fragment from pre-swizzled weights: frag[kt][nt][lane][j] dwords.
__device__ __forceinline__ v16bf bfrag_g(const unsigned* Bsw, int kt, int nt,
                                         int ntiles, int lane) {
    FragBF f;
    const unsigned* p = Bsw + ((size_t)((kt * ntiles + nt) * 32 + lane)) * 8;
#pragma unroll
    for (int j = 0; j < 8; ++j) f.u[j] = p[j];
    return f.v;
}

__device__ __forceinline__ v8f wmma_bf16(v16bf a, v16bf b, v8f c) {
    return __builtin_amdgcn_wmma_f32_16x16x32_bf16(false, a, false, b,
                                                   (short)0, c, false, false);
}

// ---------------------------------------------------------------------------
// Weight pre-swizzle: f32 [K x N] row-major -> B-fragment bf16 layout.
// out index = ((kt*ntiles + nt)*32 + lane)*8 + j ;  K(j,lane) = 2(j&3) + 16(j>>2) + 8*(lane>=16)
// ---------------------------------------------------------------------------
__global__ void swizzle_b_kernel(const float* __restrict__ W,
                                 unsigned* __restrict__ out, int K, int N) {
    int id = blockIdx.x * blockDim.x + threadIdx.x;
    int ntiles = N >> 4;
    int total = (K >> 5) * ntiles * 256;
    if (id >= total) return;
    int j = id & 7, lane = (id >> 3) & 31;
    int t = id >> 8;
    int nt = t % ntiles, kt = t / ntiles;
    int k0 = kt * 32 + 2 * (j & 3) + ((j >> 2) << 4) + (((lane >> 4) & 1) << 3);
    int n  = nt * 16 + (lane & 15);
    out[id] = pack2(W[k0 * N + n], W[(k0 + 1) * N + n]);
}

// ---------------------------------------------------------------------------
// Generic WMMA GEMM, N=128: C[M x 128] (+)= act(A[M x K] @ Bsw + bias)
// grid.x = M/16, block = 256 (8 waves, wave w handles N-subtile w).
// mode: 0 = store acc+bias, 1 = store relu(acc+bias), 2 = C += acc+bias
// ---------------------------------------------------------------------------
__global__ __launch_bounds__(256) void gemm_n128_kernel(
    const float* __restrict__ A, const unsigned* __restrict__ Bsw,
    const float* __restrict__ bias, float* __restrict__ C, int K, int mode) {
    __shared__ unsigned short sA[16][32];
    int m0 = blockIdx.x * 16;
    int lane = threadIdx.x & 31, wave = threadIdx.x >> 5;
    v8f acc = {};
    int nkt = K >> 5;
    for (int kt = 0; kt < nkt; ++kt) {
        int idx = threadIdx.x;
#pragma unroll
        for (int q = 0; q < 2; ++q, idx += 256) {
            int i = idx >> 5, k = idx & 31;
            sA[i][k] = (unsigned short)bf16u(A[(size_t)(m0 + i) * K + kt * 32 + k]);
        }
        __syncthreads();
        v16bf a = afrag_lds(&sA[0][0], 32, 0, lane);
        v16bf b = bfrag_g(Bsw, kt, wave, 8, lane);
        acc = wmma_bf16(a, b, acc);
        __syncthreads();
    }
    int n  = (wave << 4) + (lane & 15);
    int mb = ((lane >> 4) & 1) << 3;
    float bn = bias ? bias[n] : 0.0f;
#pragma unroll
    for (int r = 0; r < 8; ++r) {
        float v = acc[r] + bn;
        if (mode == 1) v = fmaxf(v, 0.0f);
        float* dst = &C[(size_t)(m0 + mb + r) * 128 + n];
        if (mode == 2) *dst += v; else *dst = v;
    }
}

// ---------------------------------------------------------------------------
// Small helpers
// ---------------------------------------------------------------------------
__global__ void zero_f_kernel(float* p, int n) {
    int i = blockIdx.x * blockDim.x + threadIdx.x;
    if (i < n) p[i] = 0.0f;
}
__global__ void init_deg_kernel(float* deg, int n) {
    int i = blockIdx.x * blockDim.x + threadIdx.x;
    if (i < n) deg[i] = 1.0f;                         // self-loop
}
__global__ void add_deg_kernel(const int* __restrict__ cols, float* deg, int e) {
    int i = blockIdx.x * blockDim.x + threadIdx.x;
    if (i < e) atomicAdd(&deg[cols[i]], 1.0f);
}
__global__ void finalize_dinv_kernel(float* deg, int n) {
    int i = blockIdx.x * blockDim.x + threadIdx.x;
    if (i < n) deg[i] = rsqrtf(deg[i]);               // deg >= 1 always
}

// node encoder layer1: t = relu(x @ ne_w1 + b1), K=10 -> scalar
__global__ void ne_layer1_kernel(const float* __restrict__ x,
                                 const float* __restrict__ w1,
                                 const float* __restrict__ b1,
                                 float* __restrict__ t) {
    int id = blockIdx.x * blockDim.x + threadIdx.x;
    if (id >= NN * HD) return;
    int node = id >> 7, f = id & 127;
    float acc = b1[f];
#pragma unroll
    for (int k = 0; k < 10; ++k) acc += x[node * 10 + k] * w1[k * 128 + f];
    t[id] = fmaxf(acc, 0.0f);
}

// GCN scatter: agg[col] += dinv[row]*dinv[col] * m[row]  (4 floats/thread)
__global__ void gcn_scatter_kernel(const int* __restrict__ rows,
                                   const int* __restrict__ cols,
                                   const float* __restrict__ dinv,
                                   const float* __restrict__ m,
                                   float* __restrict__ agg) {
    int tid = blockIdx.x * blockDim.x + threadIdx.x;
    if (tid >= NE * 32) return;
    int e = tid >> 5, f4 = (tid & 31) << 2;
    int r = rows[e], c = cols[e];
    float nm = dinv[r] * dinv[c];
    const float4 mv = *(const float4*)(m + (size_t)r * 128 + f4);
    float* dst = agg + (size_t)c * 128 + f4;
    atomicAdd(dst + 0, nm * mv.x);
    atomicAdd(dst + 1, nm * mv.y);
    atomicAdd(dst + 2, nm * mv.z);
    atomicAdd(dst + 3, nm * mv.w);
}

// self-loop contribution + bias (single writer per element, after scatter)
__global__ void gcn_self_bias_kernel(const float* __restrict__ dinv,
                                     const float* __restrict__ m,
                                     const float* __restrict__ bias,
                                     float* __restrict__ agg) {
    int id = blockIdx.x * blockDim.x + threadIdx.x;
    if (id >= NN * HD) return;
    int v = id >> 7, f = id & 127;
    float d = dinv[v];
    agg[id] += d * d * m[id] + bias[f];
}

__global__ void bn_stats_kernel(const float* __restrict__ h, float* stats,
                                int rowsPerBlock) {
    int c = threadIdx.x;                               // 128 threads
    int r0 = blockIdx.x * rowsPerBlock;
    int r1 = r0 + rowsPerBlock; if (r1 > NN) r1 = NN;
    float s = 0.0f, s2 = 0.0f;
    for (int r = r0; r < r1; ++r) {
        float v = h[(size_t)r * 128 + c];
        s += v; s2 += v * v;
    }
    atomicAdd(&stats[c], s);
    atomicAdd(&stats[128 + c], s2);
}
__global__ void bn_finalize_kernel(const float* __restrict__ stats,
                                   const float* __restrict__ g,
                                   const float* __restrict__ b, float* sc) {
    int c = threadIdx.x;
    float mu  = stats[c] * (1.0f / NN);
    float var = stats[128 + c] * (1.0f / NN) - mu * mu;
    float is  = rsqrtf(var + 1e-5f);
    float scale = g[c] * is;
    sc[c] = scale;
    sc[128 + c] = b[c] - mu * scale;
}
__global__ void bn_apply_relu_kernel(float* __restrict__ h,
                                     const float* __restrict__ sc) {
    int id = blockIdx.x * blockDim.x + threadIdx.x;
    if (id >= NN * HD) return;
    int f = id & 127;
    h[id] = fmaxf(h[id] * sc[f] + sc[128 + f], 0.0f);
}

// ---------------------------------------------------------------------------
// Fused edge head: 1 block = 16 edges, 256 threads = 8 wave32s.
// gathers h[row]/h[col] once, keeps every intermediate in LDS, WMMA for all
// K>=32 contractions, writes only the 16x2 logits.
// ---------------------------------------------------------------------------
__global__ __launch_bounds__(256) void edge_head_kernel(
    const int* __restrict__ rows, const int* __restrict__ cols,
    const float* __restrict__ edge_attr, const float* __restrict__ hfin,
    const float* __restrict__ ee_w1, const float* __restrict__ ee_b1,
    const unsigned* __restrict__ ee_w2_sw, const float* __restrict__ ee_b2,
    const unsigned* __restrict__ ea_w1_sw, const float* __restrict__ ea_w1_full,
    const float* __restrict__ ea_b1, const float* __restrict__ ea_w2,
    const float* __restrict__ ea_b2,
    const unsigned* __restrict__ cl_w1_sw, const float* __restrict__ cl_b1,
    const unsigned* __restrict__ cl_w2_sw, const float* __restrict__ cl_b2,
    const float* __restrict__ cl_w3, const float* __restrict__ cl_b3,
    float* __restrict__ out) {
    __shared__ unsigned short sHr[16][128];   // h[row]  bf16
    __shared__ unsigned short sHc[16][128];   // h[col]  bf16
    __shared__ float          sEA[16][12];    // edge_attr
    __shared__ unsigned short sT[16][128];    // relu(ee layer1) bf16
    __shared__ float          sEF[16][128];   // edge_feat f32
    __shared__ float          sAH[16][128];   // relu(att layer1) f32
    __shared__ float          sAtt[16];       // attention scalar
    __shared__ unsigned short sER[16][256];   // edge_repr bf16
    __shared__ unsigned short sZ1[16][128];   // relu(cl1) bf16
    __shared__ float          sZ2[16][64];    // relu(cl2) f32
    __shared__ int            sRow[16], sCol[16];

    int t = threadIdx.x;
    int lane = t & 31, wave = t >> 5;
    int e0 = blockIdx.x * 16;

    // -- step 1: indices, edge_attr, att accumulator init
    if (t < 16) {
        sRow[t] = rows[e0 + t];
        sCol[t] = cols[e0 + t];
        sAtt[t] = ea_b2[0];
    }
    if (t < 160) {
        int i = t / 10, k = t % 10;
        sEA[i][k] = edge_attr[(size_t)(e0 + i) * 10 + k];
    }
    __syncthreads();

    // -- step 2: gather h_r / h_c (f32 -> bf16)
    for (int idx = t; idx < 2048; idx += 256) {
        int i = idx >> 7, f = idx & 127;
        sHr[i][f] = (unsigned short)bf16u(hfin[(size_t)sRow[i] * 128 + f]);
        sHc[i][f] = (unsigned short)bf16u(hfin[(size_t)sCol[i] * 128 + f]);
    }
    // -- step 3: edge encoder layer1 (K=10, scalar)
    for (int idx = t; idx < 2048; idx += 256) {
        int i = idx >> 7, f = idx & 127;
        float acc = ee_b1[f];
#pragma unroll
        for (int k = 0; k < 10; ++k) acc += sEA[i][k] * ee_w1[k * 128 + f];
        sT[i][f] = (unsigned short)bf16u(fmaxf(acc, 0.0f));
    }
    __syncthreads();

    int n  = (wave << 4) + (lane & 15);
    int mb = ((lane >> 4) & 1) << 3;

    // -- step 4: edge encoder layer2 (K=128 WMMA), edge_feat f32 -> sEF
    {
        v8f acc = {};
#pragma unroll
        for (int kt = 0; kt < 4; ++kt)
            acc = wmma_bf16(afrag_lds(&sT[0][0], 128, kt * 32, lane),
                            bfrag_g(ee_w2_sw, kt, wave, 8, lane), acc);
        float b = ee_b2[n];
#pragma unroll
        for (int r = 0; r < 8; ++r) sEF[mb + r][n] = acc[r] + b;
    }
    // -- step 5: attention layer1 (K=256 WMMA over [h_r|h_c] + K=10 tail)
    {
        v8f acc = {};
#pragma unroll
        for (int kt = 0; kt < 4; ++kt)
            acc = wmma_bf16(afrag_lds(&sHr[0][0], 128, kt * 32, lane),
                            bfrag_g(ea_w1_sw, kt, wave, 8, lane), acc);
#pragma unroll
        for (int kt = 0; kt < 4; ++kt)
            acc = wmma_bf16(afrag_lds(&sHc[0][0], 128, kt * 32, lane),
                            bfrag_g(ea_w1_sw, 4 + kt, wave, 8, lane), acc);
        float wtail[10];
#pragma unroll
        for (int k = 0; k < 10; ++k) wtail[k] = ea_w1_full[(256 + k) * 128 + n];
        float b = ea_b1[n];
#pragma unroll
        for (int r = 0; r < 8; ++r) {
            int m = mb + r;
            float v = acc[r] + b;
#pragma unroll
            for (int k = 0; k < 10; ++k) v += sEA[m][k] * wtail[k];
            sAH[m][n] = fmaxf(v, 0.0f);
        }
    }
    __syncthreads();

    // -- step 6: attention layer2 (128-dot per edge) + sigmoid
    {
        int i = t >> 4, fo = t & 15;
        float partial = 0.0f;
#pragma unroll
        for (int f = fo; f < 128; f += 16) partial += sAH[i][f] * ea_w2[f];
        atomicAdd(&sAtt[i], partial);                  // ds_add_f32
    }
    __syncthreads();
    if (t < 16) sAtt[t] = 1.0f / (1.0f + __expf(-sAtt[t]));
    __syncthreads();

    // -- step 7: edge_repr = [h_r + att*ef, h_c + att*ef]  (bf16)
    for (int idx = t; idx < 2048; idx += 256) {
        int i = idx >> 7, f = idx & 127;
        float wef = sAtt[i] * sEF[i][f];
        sER[i][f]       = (unsigned short)bf16u(bf16f(sHr[i][f]) + wef);
        sER[i][128 + f] = (unsigned short)bf16u(bf16f(sHc[i][f]) + wef);
    }
    __syncthreads();

    // -- step 8: classifier layer1 (K=256 WMMA, N=128) -> relu -> sZ1
    {
        v8f acc = {};
#pragma unroll
        for (int kt = 0; kt < 8; ++kt)
            acc = wmma_bf16(afrag_lds(&sER[0][0], 256, kt * 32, lane),
                            bfrag_g(cl_w1_sw, kt, wave, 8, lane), acc);
        float b = cl_b1[n];
#pragma unroll
        for (int r = 0; r < 8; ++r)
            sZ1[mb + r][n] = (unsigned short)bf16u(fmaxf(acc[r] + b, 0.0f));
    }
    __syncthreads();

    // -- step 9: classifier layer2 (K=128 WMMA, N=64: waves 0..3)
    if (wave < 4) {
        v8f acc = {};
#pragma unroll
        for (int kt = 0; kt < 4; ++kt)
            acc = wmma_bf16(afrag_lds(&sZ1[0][0], 128, kt * 32, lane),
                            bfrag_g(cl_w2_sw, kt, wave, 4, lane), acc);
        float b = cl_b2[n];
#pragma unroll
        for (int r = 0; r < 8; ++r)
            sZ2[mb + r][n] = fmaxf(acc[r] + b, 0.0f);
    }
    __syncthreads();

    // -- step 10: logits (K=64, C=2, scalar)
    if (t < 32) {
        int i = t >> 1, c = t & 1;
        float v = cl_b3[c];
#pragma unroll
        for (int k = 0; k < 64; ++k) v += sZ2[i][k] * cl_w3[k * 2 + c];
        out[(size_t)(e0 + i) * 2 + c] = v;
    }
}

// ---------------------------------------------------------------------------
// Host-side launcher
// ---------------------------------------------------------------------------
extern "C" void kernel_launch(void* const* d_in, const int* in_sizes, int n_in,
                              void* d_out, int out_size, void* d_ws,
                              size_t ws_size, hipStream_t stream) {
    (void)in_sizes; (void)n_in; (void)out_size; (void)ws_size;

    const float* x         = (const float*)d_in[0];
    const int*   ei        = (const int*)d_in[1];
    const float* edge_attr = (const float*)d_in[2];
    const float* ne_w1 = (const float*)d_in[3],  *ne_b1 = (const float*)d_in[4];
    const float* ne_w2 = (const float*)d_in[5],  *ne_b2 = (const float*)d_in[6];
    const float* ee_w1 = (const float*)d_in[7],  *ee_b1 = (const float*)d_in[8];
    const float* ee_w2 = (const float*)d_in[9],  *ee_b2 = (const float*)d_in[10];
    const float* ea_w1 = (const float*)d_in[11], *ea_b1 = (const float*)d_in[12];
    const float* ea_w2 = (const float*)d_in[13], *ea_b2 = (const float*)d_in[14];
    const float* gcn_w = (const float*)d_in[15], *gcn_b = (const float*)d_in[16];
    const float* bn_g  = (const float*)d_in[17], *bn_b  = (const float*)d_in[18];
    const float* skip_w= (const float*)d_in[19], *skip_b= (const float*)d_in[20];
    const float* cl_w1 = (const float*)d_in[21], *cl_b1 = (const float*)d_in[22];
    const float* cl_w2 = (const float*)d_in[23], *cl_b2 = (const float*)d_in[24];
    const float* cl_w3 = (const float*)d_in[25], *cl_b3 = (const float*)d_in[26];

    const int* rows = ei;            // edge_index[0] (source)
    const int* cols = ei + NE;       // edge_index[1] (target)

    // workspace carve-up
    const size_t NB = (size_t)NN * HD;               // 6.4M floats per node buffer
    float* B0 = (float*)d_ws;
    float* B1 = B0 + NB;
    float* B2 = B1 + NB;
    float* B3 = B2 + NB;
    float* dinv  = B3 + NB;                          // 50000
    float* stats = dinv + 50016;                     // 256
    float* bnsc  = stats + 256;                      // 256
    unsigned* sw = (unsigned*)(bnsc + 256);
    unsigned* sw_ne2   = sw;                 // 128x128 -> 8192 dwords
    unsigned* sw_gcn0  = sw_ne2  + 8192;
    unsigned* sw_gcn1  = sw_gcn0 + 8192;
    unsigned* sw_gcn2  = sw_gcn1 + 8192;
    unsigned* sw_skip0 = sw_gcn2 + 8192;
    unsigned* sw_skip1 = sw_skip0 + 8192;
    unsigned* sw_ee2   = sw_skip1 + 8192;
    unsigned* sw_ea1   = sw_ee2  + 8192;     // 256x128 -> 16384 dwords
    unsigned* sw_cl1   = sw_ea1  + 16384;    // 256x128
    unsigned* sw_cl2   = sw_cl1  + 16384;    // 128x64 -> 4096 dwords

    // 1) weight pre-swizzle into WMMA B-fragment order
    auto swz = [&](const float* W, unsigned* O, int K, int N) {
        int total = (K >> 5) * (N >> 4) * 256;
        swizzle_b_kernel<<<(total + 255) / 256, 256, 0, stream>>>(W, O, K, N);
    };
    swz(ne_w2, sw_ne2, 128, 128);
    swz(gcn_w + 0 * 16384, sw_gcn0, 128, 128);
    swz(gcn_w + 1 * 16384, sw_gcn1, 128, 128);
    swz(gcn_w + 2 * 16384, sw_gcn2, 128, 128);
    swz(skip_w + 0 * 16384, sw_skip0, 128, 128);
    swz(skip_w + 1 * 16384, sw_skip1, 128, 128);
    swz(ee_w2, sw_ee2, 128, 128);
    swz(ea_w1, sw_ea1, 256, 128);            // first 256 rows of 266
    swz(cl_w1, sw_cl1, 256, 128);
    swz(cl_w2, sw_cl2, 128, 64);

    // 2) symmetric GCN normalization (self-loops)
    init_deg_kernel<<<(NN + 255) / 256, 256, 0, stream>>>(dinv, NN);
    add_deg_kernel<<<(NE + 255) / 256, 256, 0, stream>>>(cols, dinv, NE);
    finalize_dinv_kernel<<<(NN + 255) / 256, 256, 0, stream>>>(dinv, NN);

    // 3) node encoder: B0 = relu(x@ne_w1+b1) @ ne_w2 + b2
    ne_layer1_kernel<<<(NN * HD) / 256, 256, 0, stream>>>(x, ne_w1, ne_b1, B2);
    gemm_n128_kernel<<<NN / 16, 256, 0, stream>>>(B2, sw_ne2, ne_b2, B0, 128, 0);

    // 4) three GCN layers with buffer rotation
    const unsigned* gcn_sw[3]  = { sw_gcn0, sw_gcn1, sw_gcn2 };
    const unsigned* skip_sw[2] = { sw_skip0, sw_skip1 };
    float* hbuf[4]   = { B0, B1, B3, B0 };   // h at start of layer 0,1,2; final
    float* aggbuf[3] = { B1, B3, B0 };       // agg target per layer
    for (int i = 0; i < 3; ++i) {
        float* h   = hbuf[i];
        float* agg = aggbuf[i];
        // m = h @ gcn_w[i]
        gemm_n128_kernel<<<NN / 16, 256, 0, stream>>>(h, gcn_sw[i], nullptr, B2, 128, 0);
        // segment-sum scatter + self-loops + bias
        zero_f_kernel<<<(int)(NB / 256), 256, 0, stream>>>(agg, (int)NB);
        gcn_scatter_kernel<<<(NE * 32) / 256, 256, 0, stream>>>(rows, cols, dinv, B2, agg);
        gcn_self_bias_kernel<<<(NN * HD) / 256, 256, 0, stream>>>(dinv, B2,
                                                                  gcn_b + i * 128, agg);
        // batchnorm (batch statistics) + relu
        zero_f_kernel<<<1, 256, 0, stream>>>(stats, 256);
        bn_stats_kernel<<<(NN + 255) / 256, 128, 0, stream>>>(agg, stats, 256);
        bn_finalize_kernel<<<1, 128, 0, stream>>>(stats, bn_g + i * 128,
                                                  bn_b + i * 128, bnsc);
        bn_apply_relu_kernel<<<(NN * HD) / 256, 256, 0, stream>>>(agg, bnsc);
        // skip: agg += h_prev @ skip_w[i-1] + skip_b[i-1]
        if (i > 0)
            gemm_n128_kernel<<<NN / 16, 256, 0, stream>>>(h, skip_sw[i - 1],
                                                          skip_b + (i - 1) * 128,
                                                          agg, 128, 2);
    }
    float* hfin = B0;   // final node features

    // 5) fused edge head + classifier -> logits [NE x 2]
    edge_head_kernel<<<NE / 16, 256, 0, stream>>>(
        rows, cols, edge_attr, hfin,
        ee_w1, ee_b1, sw_ee2, ee_b2,
        sw_ea1, ea_w1, ea_b1, ea_w2, ea_b2,
        sw_cl1, cl_b1, sw_cl2, cl_b2, cl_w3, cl_b3,
        (float*)d_out);
}
